// Residual_GAT_46643344835141
// MI455X (gfx1250) — matrix operand order repaired and checked by reading, not compiled, not run on hardware
//
#include <hip/hip_runtime.h>
#include <math.h>

#define N_NODES 500000
#define N_EDGES 16000000
#define F_IN 16
#define H 4
#define NEG_SLOPE 0.2f

typedef __attribute__((ext_vector_type(16))) _Float16  v16h;
typedef __attribute__((ext_vector_type(8)))  float     v8f;
typedef __attribute__((ext_vector_type(4)))  int       v4i;
typedef __attribute__((ext_vector_type(2)))  long long v2l;

__device__ __forceinline__ float leakyf(float v) { return v > 0.f ? v : NEG_SLOPE * v; }

// Composite float atomic-max trick (correct for mixed signs).
__device__ __forceinline__ void atomicMaxF(float* addr, float v) {
    if (v >= 0.f) atomicMax((int*)addr, __float_as_int(v));
    else          atomicMin((unsigned int*)addr, __float_as_uint(v));
}

// ---------------------------------------------------------------------------
// 0. Pooling accumulators: [0..3]=sum, [4..7]=max
__global__ void k_pool_init(float* pools) {
    int t = threadIdx.x;
    if (t < H) pools[t] = 0.f;
    else if (t < 2 * H) pools[t] = -3.4e38f;
}

// ---------------------------------------------------------------------------
// 1. Compact int64 edge_index -> int32 src/dst.
//    int64 stream is read ONCE -> non-temporal loads so the int32 arrays
//    (128 MB) + node arrays (~36 MB) stay resident in the 192 MB L2 for the
//    four subsequent edge passes. 4 edges per thread, b128 traffic.
__global__ void k_convert(const long long* __restrict__ ei,
                          int* __restrict__ s32, int* __restrict__ d32) {
    long long e = (long long)(blockIdx.x * blockDim.x + threadIdx.x) * 4;
    if (e >= N_EDGES) return;
    v2l sa = __builtin_nontemporal_load((const v2l*)(ei + e));
    v2l sb = __builtin_nontemporal_load((const v2l*)(ei + e + 2));
    v2l da = __builtin_nontemporal_load((const v2l*)(ei + (long long)N_EDGES + e));
    v2l db = __builtin_nontemporal_load((const v2l*)(ei + (long long)N_EDGES + e + 2));
    v4i sv, dv;
    sv[0] = (int)sa[0]; sv[1] = (int)sa[1]; sv[2] = (int)sb[0]; sv[3] = (int)sb[1];
    dv[0] = (int)da[0]; dv[1] = (int)da[1]; dv[2] = (int)db[0]; dv[3] = (int)db[1];
    *(v4i*)(s32 + e) = sv;      // regular-temporal: keep hot in L2
    *(v4i*)(d32 + e) = dv;
}

// ---------------------------------------------------------------------------
// 2. Layer-1 combined B (f16, 32x16 K-major):
//    cols 0..3 = W1, cols 4..7 = res_W, col 8 = W1@a_src1, col 9 = W1@a_dst1
__global__ void k_prepB1(const float* __restrict__ W1, const float* __restrict__ a_src,
                         const float* __restrict__ a_dst, const float* __restrict__ resW,
                         unsigned short* __restrict__ Braw) {
    _Float16* B = (_Float16*)Braw;
    int t = threadIdx.x;            // 512 threads cover 32x16
    int k = t >> 4, n = t & 15;
    float v = 0.f;
    if (k < F_IN) {
        if (n < H)           v = W1[k * H + n];
        else if (n < 2 * H)  v = resW[k * H + (n - H)];
        else if (n == 8) { for (int j = 0; j < H; ++j) v += W1[k * H + j] * a_src[j]; }
        else if (n == 9) { for (int j = 0; j < H; ++j) v += W1[k * H + j] * a_dst[j]; }
    }
    B[t] = (_Float16)v;
}

// ---------------------------------------------------------------------------
// SE gate math (shared by both channel-attention stages)
__device__ void compute_gate(const float* pools, const float* w1, const float* w2, float* g) {
    float avg[H], mx[H];
    for (int c = 0; c < H; ++c) { avg[c] = pools[c] * (1.f / (float)N_NODES); mx[c] = pools[H + c]; }
    float mida[H / 2], midm[H / 2];
    for (int j = 0; j < H / 2; ++j) {
        float sa = 0.f, sm = 0.f;
        for (int c = 0; c < H; ++c) { sa += avg[c] * w1[c * (H / 2) + j]; sm += mx[c] * w1[c * (H / 2) + j]; }
        mida[j] = fmaxf(sa, 0.f); midm[j] = fmaxf(sm, 0.f);
    }
    for (int n = 0; n < H; ++n) {
        float s = 0.f;
        for (int j = 0; j < H / 2; ++j) s += (mida[j] + midm[j]) * w2[j * H + n];
        g[n] = 1.f / (1.f + __expf(-s));
    }
}

// 3. CA1 gate + layer-2 combined B with the gate folded into W2's rows
//    (h*gate @ W2 == h @ diag(gate)W2). Also resets pools for layer 2.
__global__ void k_gate_prepB2(float* __restrict__ pools,
                              const float* __restrict__ caw1, const float* __restrict__ caw2,
                              const float* __restrict__ W2, const float* __restrict__ a_src,
                              const float* __restrict__ a_dst, unsigned short* __restrict__ Braw) {
    __shared__ float g[H];
    int t = threadIdx.x;
    if (t == 0) {
        float gg[H];
        compute_gate(pools, caw1, caw2, gg);
        for (int c = 0; c < H; ++c) g[c] = gg[c];
        for (int c = 0; c < H; ++c) { pools[c] = 0.f; pools[H + c] = -3.4e38f; }
    }
    __syncthreads();
    _Float16* B = (_Float16*)Braw;
    int k = t >> 4, n = t & 15;
    float v = 0.f;
    if (k < H) {
        if (n < H) v = g[k] * W2[k * H + n];
        else if (n == 8) { float s = 0.f; for (int j = 0; j < H; ++j) s += W2[k * H + j] * a_src[j]; v = g[k] * s; }
        else if (n == 9) { float s = 0.f; for (int j = 0; j < H; ++j) s += W2[k * H + j] * a_dst[j]; v = g[k] * s; }
    }
    B[t] = (_Float16)v;
}

// ---------------------------------------------------------------------------
// 4. Fused node transform: one v_wmma_f32_16x16x32_f16 per 16-node tile.
//    D cols 0..3 -> hout (+acc zeroing), 4..7 -> residual(+res_b),
//    8 -> as (+m seed, denom zero via lane+1 shuffle), 9 -> ad.
//    Replaces the separate node-init pass entirely.
template <bool HAS_RES>
__global__ void k_wmma_transform(const float* __restrict__ X, int Kin,
                                 const unsigned short* __restrict__ Braw,
                                 float* __restrict__ hout,
                                 float* __restrict__ resout, const float* __restrict__ res_b,
                                 float* __restrict__ as_out, float* __restrict__ ad_out,
                                 float* __restrict__ m_out, float* __restrict__ denom,
                                 float* __restrict__ acc) {
    const _Float16* B = (const _Float16*)Braw;
    int wave = (blockIdx.x * blockDim.x + threadIdx.x) >> 5;   // wave32
    int lane = threadIdx.x & 31;
    int base = wave * 16;
    if (base >= N_NODES) return;            // wave-uniform exit (EXEC all-1s for WMMA)

    int  m  = lane & 15;
    bool hi = lane >= 16;

    // A: 16x32 f16; lanes 0-15 hold K 0..7 (halves 0-7) & K 16..23 (zeros);
    //    lanes 16-31 hold K 8..15 & K 24..31 (zeros).
    int kbase = hi ? 8 : 0;
    const float* xr = X + (size_t)(base + m) * Kin;
    v16h a, b;
#pragma unroll
    for (int t = 0; t < 8; ++t) {
        int k = kbase + t;
        a[t]     = (_Float16)((k < Kin) ? xr[k] : 0.f);
        a[t + 8] = (_Float16)0.f;
    }
    // B: 32x16 f16; lanes 0-15 = K 0..7 & 16..23, lanes 16-31 = K 8..15 & 24..31; column = lane%16.
#pragma unroll
    for (int t = 0; t < 16; ++t) b[t] = B[((t < 8 ? 0 : 16) + kbase + (t & 7)) * 16 + m];

    v8f c = {0.f, 0.f, 0.f, 0.f, 0.f, 0.f, 0.f, 0.f};
    c = __builtin_amdgcn_wmma_f32_16x16x32_f16(false, a, false, b, (short)0, c, false, false);

    float rb = 0.f;
    if (HAS_RES && m >= H && m < 2 * H) rb = res_b[m - H];

    // D layout: VGPR r holds M=r (lanes 0-15) / M=r+8 (lanes 16-31), N = lane%16.
#pragma unroll
    for (int r = 0; r < 8; ++r) {
        int node = base + r + (hi ? 8 : 0);
        float v = c[r];
        float vnext = __shfl(v, lane + 1);  // lane 8/24: ad partner of as
        if (m < H) {
            hout[(size_t)node * H + m] = v;
            acc[(size_t)node * H + m] = 0.f;
        } else if (HAS_RES && m < 2 * H) {
            resout[(size_t)node * H + (m - H)] = v + rb;
        } else if (m == 8) {
            as_out[node] = v;
            m_out[node]  = leakyf(v + vnext);   // self-loop seeds segment-max
            denom[node]  = 0.f;
        } else if (m == 9) {
            ad_out[node] = v;
        }
    }
}

// ---------------------------------------------------------------------------
// 5. Edge pass 1: segment-max of attention logits. 4 edges/thread (b128 idx).
__global__ void k_edge_max(const int* __restrict__ s32, const int* __restrict__ d32,
                           const float* __restrict__ as, const float* __restrict__ ad,
                           float* __restrict__ m) {
    long long e = (long long)(blockIdx.x * blockDim.x + threadIdx.x) * 4;
    if (e >= N_EDGES) return;
    v4i sv = *(const v4i*)(s32 + e);
    v4i dv = *(const v4i*)(d32 + e);
#pragma unroll
    for (int j = 0; j < 4; ++j) {
        int s = sv[j], d = dv[j];
        atomicMaxF(&m[d], leakyf(as[s] + ad[d]));
    }
}

// 6. Edge pass 2: fused exp + denominator + weighted feature scatter.
__global__ void k_edge_acc(const int* __restrict__ s32, const int* __restrict__ d32,
                           const float* __restrict__ as, const float* __restrict__ ad,
                           const float* __restrict__ m, const float* __restrict__ h,
                           float* __restrict__ denom, float* __restrict__ acc) {
    long long e = (long long)(blockIdx.x * blockDim.x + threadIdx.x) * 4;
    if (e >= N_EDGES) return;
    v4i sv = *(const v4i*)(s32 + e);
    v4i dv = *(const v4i*)(d32 + e);
#pragma unroll
    for (int j = 0; j < 4; ++j) {
        int s = sv[j], d = dv[j];
        float ex = __expf(leakyf(as[s] + ad[d]) - m[d]);
        atomicAdd(&denom[d], ex);
        const float4 hv = ((const float4*)h)[s];
        atomicAdd(&acc[(size_t)d * 4 + 0], ex * hv.x);
        atomicAdd(&acc[(size_t)d * 4 + 1], ex * hv.y);
        atomicAdd(&acc[(size_t)d * 4 + 2], ex * hv.z);
        atomicAdd(&acc[(size_t)d * 4 + 3], ex * hv.w);
    }
}

// 7. Finalize: add self-loop term, normalize, +bias, ReLU; fused SE pooling.
__global__ void k_finalize(const float* __restrict__ as, const float* __restrict__ ad,
                           const float* __restrict__ m, const float* __restrict__ denom,
                           const float* __restrict__ acc, const float* __restrict__ h,
                           const float* __restrict__ bias, float* __restrict__ hrelu,
                           float* __restrict__ pools) {
    __shared__ float ssum[H];
    __shared__ float smax[H];
    int tid = threadIdx.x;
    if (tid < H) { ssum[tid] = 0.f; smax[tid] = -3.4e38f; }
    __syncthreads();

    int i = blockIdx.x * blockDim.x + tid;
    if (i < N_NODES) {
        float ex = __expf(leakyf(as[i] + ad[i]) - m[i]);
        float inv = 1.f / (denom[i] + ex);
        float4 hv = ((const float4*)h)[i];
        float hc[H] = {hv.x, hv.y, hv.z, hv.w};
        float o[H];
        for (int c = 0; c < H; ++c) {
            float v = (acc[(size_t)i * 4 + c] + ex * hc[c]) * inv + bias[c];
            v = fmaxf(v, 0.f);
            o[c] = v;
            atomicAdd(&ssum[c], v);
            atomicMaxF(&smax[c], v);
        }
        float4 ov = {o[0], o[1], o[2], o[3]};
        ((float4*)hrelu)[i] = ov;
    }
    __syncthreads();
    if (tid < H) {
        atomicAdd(&pools[tid], ssum[tid]);
        atomicMaxF(&pools[H + tid], smax[tid]);
    }
}

// 8. CA2 gate (applied in the final kernel).
__global__ void k_gate2(const float* __restrict__ pools, const float* __restrict__ caw1,
                        const float* __restrict__ caw2, float* __restrict__ gate) {
    if (threadIdx.x == 0) {
        float g[H];
        compute_gate(pools, caw1, caw2, g);
        for (int c = 0; c < H; ++c) gate[c] = g[c];
    }
}

// 9. Final: h2*gate2 + residual -> fc -> sigmoid.
__global__ void k_final(const float* __restrict__ hrelu, const float* __restrict__ resid,
                        const float* __restrict__ gate, const float* __restrict__ fcW,
                        const float* __restrict__ fcb, float* __restrict__ out) {
    int i = blockIdx.x * blockDim.x + threadIdx.x;
    if (i >= N_NODES) return;
    float4 hv = ((const float4*)hrelu)[i];
    float4 rv = ((const float4*)resid)[i];
    float hc[H] = {hv.x, hv.y, hv.z, hv.w};
    float rc[H] = {rv.x, rv.y, rv.z, rv.w};
    float v = fcb[0];
    for (int c = 0; c < H; ++c) v += (hc[c] * gate[c] + rc[c]) * fcW[c];
    out[i] = 1.f / (1.f + __expf(-v));
}

// ---------------------------------------------------------------------------
extern "C" void kernel_launch(void* const* d_in, const int* in_sizes, int n_in,
                              void* d_out, int out_size, void* d_ws, size_t ws_size,
                              hipStream_t stream) {
    (void)in_sizes; (void)n_in; (void)out_size; (void)ws_size;

    const float*     x       = (const float*)d_in[0];
    const long long* eidx    = (const long long*)d_in[1];
    const float*     W1      = (const float*)d_in[2];
    const float*     a_src1  = (const float*)d_in[3];
    const float*     a_dst1  = (const float*)d_in[4];
    const float*     b1      = (const float*)d_in[5];
    const float*     W2      = (const float*)d_in[6];
    const float*     a_src2  = (const float*)d_in[7];
    const float*     a_dst2  = (const float*)d_in[8];
    const float*     b2      = (const float*)d_in[9];
    const float*     ca1_w1  = (const float*)d_in[10];
    const float*     ca1_w2  = (const float*)d_in[11];
    const float*     ca2_w1  = (const float*)d_in[12];
    const float*     ca2_w2  = (const float*)d_in[13];
    const float*     res_W   = (const float*)d_in[14];
    const float*     res_b   = (const float*)d_in[15];
    const float*     fc_W    = (const float*)d_in[16];
    const float*     fc_b    = (const float*)d_in[17];
    float*           out     = (float*)d_out;

    // Workspace carve-up (all re-initialized every call; ws is poisoned once).
    char*  w   = (char*)d_ws;
    size_t off = 0;
    auto alloc = [&](size_t bytes) -> void* {
        void* p = w + off;
        off += (bytes + 255) & ~(size_t)255;
        return p;
    };
    int*            src32 = (int*)alloc((size_t)N_EDGES * 4);
    int*            dst32 = (int*)alloc((size_t)N_EDGES * 4);
    float*          hfeat = (float*)alloc((size_t)N_NODES * H * 4);
    float*          hrelu = (float*)alloc((size_t)N_NODES * H * 4);
    float*          resid = (float*)alloc((size_t)N_NODES * H * 4);
    float*          as_b  = (float*)alloc((size_t)N_NODES * 4);
    float*          ad_b  = (float*)alloc((size_t)N_NODES * 4);
    float*          m_b   = (float*)alloc((size_t)N_NODES * 4);
    float*          den_b = (float*)alloc((size_t)N_NODES * 4);
    float*          acc_b = (float*)alloc((size_t)N_NODES * H * 4);
    float*          pools = (float*)alloc(2 * H * 4);
    float*          gate2 = (float*)alloc(H * 4);
    unsigned short* B1    = (unsigned short*)alloc(32 * 16 * 2);
    unsigned short* B2    = (unsigned short*)alloc(32 * 16 * 2);

    const int TB = 256;
    const int edge4Blocks = (N_EDGES / 4 + TB - 1) / TB;
    const int nodeBlocks  = (N_NODES + TB - 1) / TB;
    const int tiles       = (N_NODES + 15) / 16;
    const int wmmaBlocks  = (tiles + (TB / 32) - 1) / (TB / 32);

    // ---- setup ----
    k_pool_init<<<1, 32, 0, stream>>>(pools);
    k_convert<<<edge4Blocks, TB, 0, stream>>>(eidx, src32, dst32);
    k_prepB1<<<1, 512, 0, stream>>>(W1, a_src1, a_dst1, res_W, B1);

    // ---- GAT layer 1 (fused residual projection + node init) ----
    k_wmma_transform<true><<<wmmaBlocks, TB, 0, stream>>>(
        x, F_IN, B1, hfeat, resid, res_b, as_b, ad_b, m_b, den_b, acc_b);
    k_edge_max<<<edge4Blocks, TB, 0, stream>>>(src32, dst32, as_b, ad_b, m_b);
    k_edge_acc<<<edge4Blocks, TB, 0, stream>>>(src32, dst32, as_b, ad_b, m_b, hfeat, den_b, acc_b);
    k_finalize<<<nodeBlocks, TB, 0, stream>>>(as_b, ad_b, m_b, den_b, acc_b, hfeat, b1, hrelu, pools);

    // ---- CA1 gate folded into layer-2 weights ----
    k_gate_prepB2<<<1, 512, 0, stream>>>(pools, ca1_w1, ca1_w2, W2, a_src2, a_dst2, B2);

    // ---- GAT layer 2 (fused node init) ----
    k_wmma_transform<false><<<wmmaBlocks, TB, 0, stream>>>(
        hrelu, H, B2, hfeat, nullptr, nullptr, as_b, ad_b, m_b, den_b, acc_b);
    k_edge_max<<<edge4Blocks, TB, 0, stream>>>(src32, dst32, as_b, ad_b, m_b);
    k_edge_acc<<<edge4Blocks, TB, 0, stream>>>(src32, dst32, as_b, ad_b, m_b, hfeat, den_b, acc_b);
    k_finalize<<<nodeBlocks, TB, 0, stream>>>(as_b, ad_b, m_b, den_b, acc_b, hfeat, b2, hrelu, pools);

    // ---- CA2 + residual + head ----
    k_gate2<<<1, 32, 0, stream>>>(pools, ca2_w1, ca2_w2, gate2);
    k_final<<<nodeBlocks, TB, 0, stream>>>(hrelu, resid, gate2, fc_W, fc_b, out);
}
